// GINNet_80633716015165
// MI455X (gfx1250) — compile-verified, matrix operand order checked by reading, not run on hardware
//
#include <hip/hip_runtime.h>

// GIN forward, MI455X (gfx1250, wave32, WMMA).
// Roofline: edge-head GEMMs dominate (105 GFLOP) but the per-edge row gather
// dominates bytes (~0.8 GB in f16) -> gather-bandwidth bound; so node states
// are kept in f16 and all big GEMMs use v_wmma_f32_16x16x32_f16 (fp32 accum).

#define N_NODES 20000
#define N_EDGES 320000
#define HID     128
#define NLAYERS 4
#define BN_EPS  1e-5f

typedef __attribute__((ext_vector_type(16))) _Float16 v16h;
typedef __attribute__((ext_vector_type(8)))  _Float16 v8h;
typedef __attribute__((ext_vector_type(8)))  float    v8f;

// ---------------------------------------------------------------------------
// A-fragment loader: 16x32 f16 A-matrix, row-major source row.
// ISA layout (CDNA5 ISA): lanes 0-15 hold K={kb..kb+7, kb+16..kb+23},
// lanes 16-31 hold K={kb+8..kb+15, kb+24..kb+31}, M = lane%16.
// ---------------------------------------------------------------------------
__device__ __forceinline__ v16h load_a_frag(const _Float16* rowp, int kb, int lane) {
    const int klo = (lane & 16) ? 8 : 0;
    v8h lo = *(const v8h*)(rowp + kb + klo);        // 16B aligned
    v8h hi = *(const v8h*)(rowp + kb + 16 + klo);   // 16B aligned
    v16h a;
#pragma unroll
    for (int i = 0; i < 8; ++i) { a[i] = lo[i]; a[i + 8] = hi[i]; }
    return a;
}

// ---------------------------------------------------------------------------
// Pack fp32 weight W[Ktot][Ntot] (row-major) into per-(ktile,ntile) B-fragment
// blocks: block = (kt*NT+nt), within block lane-major 32 lanes x 16 halves.
// Lane l supplies column n=l%16; element e maps to K as in load_a_frag.
// ---------------------------------------------------------------------------
__global__ void pack_b_kernel(const float* __restrict__ W, _Float16* __restrict__ out,
                              int Ktot, int Ntot) {
    int tid = blockIdx.x * blockDim.x + threadIdx.x;
    int total = Ktot * Ntot;
    if (tid >= total) return;
    int e = tid & 15;
    int t = tid >> 4;
    int l = t & 31;  t >>= 5;
    int NT = Ntot >> 4;
    int nt = t % NT;
    int kt = t / NT;
    int n   = l & 15;
    int klo = (l & 16) ? 8 : 0;
    int k   = kt * 32 + ((e < 8) ? (klo + e) : (16 + klo + (e - 8)));
    out[tid] = (_Float16)W[k * Ntot + nt * 16 + n];
}

// ---------------------------------------------------------------------------
// x = h @ emb_w + emb_b   (IN_DIM=2, trivial)
// ---------------------------------------------------------------------------
__global__ void embed_kernel(const float* __restrict__ h, const float* __restrict__ ew,
                             const float* __restrict__ eb,
                             float* __restrict__ X, _Float16* __restrict__ Xh) {
    int i = blockIdx.x * blockDim.x + threadIdx.x;
    if (i >= N_NODES * HID) return;
    int n = i >> 7, c = i & 127;
    float v = h[n * 2 + 0] * ew[c] + h[n * 2 + 1] * ew[HID + c] + eb[c];
    X[i] = v;
    Xh[i] = (_Float16)v;
}

__global__ void zero_f32_kernel(float* __restrict__ p, int n) {
    int i = blockIdx.x * blockDim.x + threadIdx.x;
    if (i < n) p[i] = 0.f;
}

// segment_sum(x[src], dst): one edge per 32 threads, 4 columns per thread.
__global__ void scatter_add_kernel(const float* __restrict__ X, const int* __restrict__ src,
                                   const int* __restrict__ dst, float* __restrict__ neigh) {
    int t = blockIdx.x * blockDim.x + threadIdx.x;
    int e = t >> 5;
    if (e >= N_EDGES) return;
    int c0 = (t & 31) * 4;
    const float* xr = X + (size_t)src[e] * HID + c0;
    float*       nr = neigh + (size_t)dst[e] * HID + c0;
    float4 v = *(const float4*)xr;
    atomicAdd(nr + 0, v.x); atomicAdd(nr + 1, v.y);
    atomicAdd(nr + 2, v.z); atomicAdd(nr + 3, v.w);
}

// t = (1+eps[l])*x + neigh  -> f16
__global__ void combine_kernel(const float* __restrict__ X, const float* __restrict__ neigh,
                               const float* __restrict__ eps, int layer,
                               _Float16* __restrict__ T16) {
    int i = blockIdx.x * blockDim.x + threadIdx.x;
    if (i >= N_NODES * HID) return;
    T16[i] = (_Float16)((1.f + eps[layer]) * X[i] + neigh[i]);
}

// ---------------------------------------------------------------------------
// Node GEMM: out[M,128] = A[M,128](f16) @ Bp(packed f16, K=128) + bias, fp32 out.
// One wave per 16-row tile; 4 K-steps x 8 N-tiles of v_wmma_f32_16x16x32_f16.
// ---------------------------------------------------------------------------
__global__ void node_gemm_kernel(const _Float16* __restrict__ A, const _Float16* __restrict__ Bp,
                                 const float* __restrict__ bias, float* __restrict__ out,
                                 int ntiles) {
    int wave = threadIdx.x >> 5;
    int lane = threadIdx.x & 31;
    int mt = blockIdx.x * (blockDim.x >> 5) + wave;
    if (mt >= ntiles) return;   // whole-wave uniform: EXEC stays all-ones for WMMA
    const _Float16* rowp = A + (size_t)(mt * 16 + (lane & 15)) * HID;
    v8f c[8];
    v8f zero = {0.f, 0.f, 0.f, 0.f, 0.f, 0.f, 0.f, 0.f};
#pragma unroll
    for (int nt = 0; nt < 8; ++nt) c[nt] = zero;
#pragma unroll
    for (int kt = 0; kt < 4; ++kt) {
        v16h a = load_a_frag(rowp, kt * 32, lane);
#pragma unroll
        for (int nt = 0; nt < 8; ++nt) {
            v16h b = *(const v16h*)(Bp + (size_t)((kt * 8 + nt) * 32 + lane) * 16);
            c[nt] = __builtin_amdgcn_wmma_f32_16x16x32_f16(
                false, a, false, b, (short)0, c[nt], false, false);
        }
    }
    // C/D layout: vgpr v, lanes 0-15 -> M=v, lanes 16-31 -> M=8+v, N=lane%16
    int mbase = mt * 16 + 8 * (lane >> 4);
    int ncol  = lane & 15;
#pragma unroll
    for (int nt = 0; nt < 8; ++nt) {
        int col = nt * 16 + ncol;
        float bb = bias[col];
#pragma unroll
        for (int v = 0; v < 8; ++v)
            out[(size_t)(mbase + v) * HID + col] = c[nt][v] + bb;
    }
}

// Per-column sum / sum-of-squares over rows (for BatchNorm batch stats).
__global__ void col_stats_kernel(const float* __restrict__ U, float* __restrict__ colsum,
                                 float* __restrict__ colsq, int nrows) {
    int c = threadIdx.x;   // 128 threads
    float s = 0.f, sq = 0.f;
    for (int r = blockIdx.x; r < nrows; r += gridDim.x) {
        float v = U[(size_t)r * HID + c];
        s += v; sq += v * v;
    }
    atomicAdd(colsum + c, s);
    atomicAdd(colsq + c, sq);
}

// mean / rsqrt(var+eps) from sums (biased variance, matching reference).
__global__ void finalize_stats_kernel(const float* __restrict__ colsum,
                                      const float* __restrict__ colsq,
                                      float invN, float* __restrict__ mi) {
    int c = threadIdx.x;   // 128 threads
    float m = colsum[c] * invN;
    float var = colsq[c] * invN - m * m;
    mi[c]       = m;
    mi[HID + c] = rsqrtf(var + BN_EPS);
}

// relu(bn(U)) -> f16
__global__ void bn_relu_f16_kernel(const float* __restrict__ U, const float* __restrict__ mi,
                                   const float* __restrict__ g, const float* __restrict__ b,
                                   _Float16* __restrict__ V) {
    int i = blockIdx.x * blockDim.x + threadIdx.x;
    if (i >= N_NODES * HID) return;
    int c = i & 127;
    float v = (U[i] - mi[c]) * mi[HID + c] * g[c] + b[c];
    V[i] = (_Float16)fmaxf(v, 0.f);
}

// relu(bn(U)) -> f32
__global__ void bn_relu_f32_kernel(const float* __restrict__ U, const float* __restrict__ mi,
                                   const float* __restrict__ g, const float* __restrict__ b,
                                   float* __restrict__ Y) {
    int i = blockIdx.x * blockDim.x + threadIdx.x;
    if (i >= N_NODES * HID) return;
    int c = i & 127;
    float v = (U[i] - mi[c]) * mi[HID + c] * g[c] + b[c];
    Y[i] = fmaxf(v, 0.f);
}

// x = x + relu(bn(Y1)); refresh both fp32 and f16 copies of x.
__global__ void gin_update_kernel(const float* __restrict__ Y1, const float* __restrict__ mi,
                                  const float* __restrict__ g, const float* __restrict__ b,
                                  float* __restrict__ X, _Float16* __restrict__ Xh) {
    int i = blockIdx.x * blockDim.x + threadIdx.x;
    if (i >= N_NODES * HID) return;
    int c = i & 127;
    float v = (Y1[i] - mi[c]) * mi[HID + c] * g[c] + b[c];
    float nx = X[i] + fmaxf(v, 0.f);
    X[i] = nx;
    Xh[i] = (_Float16)nx;
}

// ---------------------------------------------------------------------------
// Edge head: ef = concat(x[src],x[dst]) [16,256]; z = relu(ef@W1+b1) via WMMA
// (8 K-steps: 4 from src row, 4 from dst row); then z@W2+b2 (128->2) via LDS.
// One wave per 16-edge tile, 4 waves / block. E/16 = 20000 tiles, 5000 blocks.
// ---------------------------------------------------------------------------
__global__ void edge_predict_kernel(const _Float16* __restrict__ Xh,
                                    const int* __restrict__ src, const int* __restrict__ dst,
                                    const _Float16* __restrict__ B1p,
                                    const float* __restrict__ b1,
                                    const float* __restrict__ W2, const float* __restrict__ b2,
                                    float* __restrict__ out, int accumulate) {
    __shared__ float zs[4][16 * HID];   // 32 KB
    int wave = threadIdx.x >> 5;
    int lane = threadIdx.x & 31;
    int mt = blockIdx.x * 4 + wave;
    int ebase = mt * 16;
    int e = ebase + (lane & 15);
    const _Float16* srow = Xh + (size_t)src[e] * HID;
    const _Float16* drow = Xh + (size_t)dst[e] * HID;

    v8f c[8];
    v8f zero = {0.f, 0.f, 0.f, 0.f, 0.f, 0.f, 0.f, 0.f};
#pragma unroll
    for (int nt = 0; nt < 8; ++nt) c[nt] = zero;

#pragma unroll
    for (int kt = 0; kt < 8; ++kt) {
        const _Float16* rowp = (kt < 4) ? srow : drow;
        int kb = (kt < 4) ? kt * 32 : (kt - 4) * 32;
        v16h a = load_a_frag(rowp, kb, lane);
#pragma unroll
        for (int nt = 0; nt < 8; ++nt) {
            v16h b = *(const v16h*)(B1p + (size_t)((kt * 8 + nt) * 32 + lane) * 16);
            c[nt] = __builtin_amdgcn_wmma_f32_16x16x32_f16(
                false, a, false, b, (short)0, c[nt], false, false);
        }
    }

    // bias + relu -> LDS staging tile (16 edges x 128)
    int mrow = 8 * (lane >> 4);
    int ncol = lane & 15;
#pragma unroll
    for (int nt = 0; nt < 8; ++nt) {
        int col = nt * 16 + ncol;
        float bb = b1[col];
#pragma unroll
        for (int v = 0; v < 8; ++v) {
            float val = c[nt][v] + bb;
            zs[wave][(mrow + v) * HID + col] = fmaxf(val, 0.f);
        }
    }
    __syncthreads();

    // second linear: 128 -> 2.  32 lanes = 16 edges x 2 classes.
    int le = lane >> 1, cl = lane & 1;
    float acc = b2[cl];
    const float* zrow = &zs[wave][le * HID];
#pragma unroll 8
    for (int j = 0; j < HID; ++j) acc += zrow[j] * W2[j * 2 + cl];
    int eo = (ebase + le) * 2 + cl;
    if (accumulate) out[eo] += acc;
    else            out[eo]  = acc;
}

// ---------------------------------------------------------------------------
// Host-side orchestration
// ---------------------------------------------------------------------------
extern "C" void kernel_launch(void* const* d_in, const int* in_sizes, int n_in,
                              void* d_out, int out_size, void* d_ws, size_t ws_size,
                              hipStream_t stream) {
    (void)in_sizes; (void)n_in; (void)out_size; (void)ws_size;

    const float* h        = (const float*)d_in[0];
    const int*   src      = (const int*)  d_in[1];
    const int*   dst      = (const int*)  d_in[2];
    const float* emb_w    = (const float*)d_in[3];
    const float* emb_b    = (const float*)d_in[4];
    const float* eps      = (const float*)d_in[5];
    const float* mlp_w1   = (const float*)d_in[6];
    const float* mlp_b1   = (const float*)d_in[7];
    const float* mlp_bn_g = (const float*)d_in[8];
    const float* mlp_bn_b = (const float*)d_in[9];
    const float* mlp_w2   = (const float*)d_in[10];
    const float* mlp_b2   = (const float*)d_in[11];
    const float* app_bn_g = (const float*)d_in[12];
    const float* app_bn_b = (const float*)d_in[13];
    const float* gin_bn_g = (const float*)d_in[14];
    const float* gin_bn_b = (const float*)d_in[15];
    const float* pred_w1  = (const float*)d_in[16];
    const float* pred_b1  = (const float*)d_in[17];
    const float* pred_w2  = (const float*)d_in[18];
    const float* pred_b2  = (const float*)d_in[19];
    float* out = (float*)d_out;

    // ---- workspace bump allocator (all offsets 256B aligned) ----
    char* ws = (char*)d_ws;
    size_t off = 0;
    auto alloc = [&](size_t bytes) -> void* {
        void* p = ws + off;
        off += (bytes + 255) & ~(size_t)255;
        return p;
    };
    const size_t NH = (size_t)N_NODES * HID;
    float*     X     = (float*)    alloc(NH * 4);
    _Float16*  Xh    = (_Float16*) alloc(NH * 2);
    float*     neigh = (float*)    alloc(NH * 4);
    _Float16*  T16   = (_Float16*) alloc(NH * 2);
    float*     U     = (float*)    alloc(NH * 4);
    _Float16*  V16   = (_Float16*) alloc(NH * 2);
    float*     Wb    = (float*)    alloc(NH * 4);
    float*     Y1    = (float*)    alloc(NH * 4);
    float*     stats = (float*)    alloc(256 * 4);  // [colsum|colsq]
    float*     mi    = (float*)    alloc(256 * 4);  // [mean|invstd]
    _Float16*  w1p   = (_Float16*) alloc((size_t)NLAYERS * HID * HID * 2);
    _Float16*  w2p   = (_Float16*) alloc((size_t)NLAYERS * HID * HID * 2);
    _Float16*  p1p   = (_Float16*) alloc((size_t)(NLAYERS + 1) * 2 * HID * HID * 2);

    const int TPB = 256;
    const int nh_blocks = (int)((NH + TPB - 1) / TPB);
    const int mlp_elems  = HID * HID;        // 16384
    const int pred_elems = 2 * HID * HID;    // 32768
    const int node_tiles = N_NODES / 16;     // 1250
    const int edge_blocks = N_EDGES / 64;    // 5000 (4 waves x 16 edges)

    // ---- pack all WMMA B-operands (f16, fragment layout) ----
    for (int l = 0; l < NLAYERS; ++l) {
        pack_b_kernel<<<(mlp_elems + TPB - 1) / TPB, TPB, 0, stream>>>(
            mlp_w1 + (size_t)l * mlp_elems, w1p + (size_t)l * mlp_elems, HID, HID);
        pack_b_kernel<<<(mlp_elems + TPB - 1) / TPB, TPB, 0, stream>>>(
            mlp_w2 + (size_t)l * mlp_elems, w2p + (size_t)l * mlp_elems, HID, HID);
    }
    for (int i = 0; i <= NLAYERS; ++i) {
        pack_b_kernel<<<(pred_elems + TPB - 1) / TPB, TPB, 0, stream>>>(
            pred_w1 + (size_t)i * pred_elems, p1p + (size_t)i * pred_elems, 2 * HID, HID);
    }

    // ---- embedding + head 0 ----
    embed_kernel<<<nh_blocks, TPB, 0, stream>>>(h, emb_w, emb_b, X, Xh);
    edge_predict_kernel<<<edge_blocks, 128, 0, stream>>>(
        Xh, src, dst, p1p, pred_b1, pred_w2, pred_b2, out, /*accumulate=*/0);

    // ---- GIN layers ----
    for (int l = 0; l < NLAYERS; ++l) {
        zero_f32_kernel<<<nh_blocks, TPB, 0, stream>>>(neigh, (int)NH);
        scatter_add_kernel<<<(N_EDGES * 32 + TPB - 1) / TPB, TPB, 0, stream>>>(X, src, dst, neigh);
        combine_kernel<<<nh_blocks, TPB, 0, stream>>>(X, neigh, eps, l, T16);

        // MLP linear 1 + BN + ReLU
        node_gemm_kernel<<<(node_tiles + 3) / 4, 128, 0, stream>>>(
            T16, w1p + (size_t)l * mlp_elems, mlp_b1 + (size_t)l * HID, U, node_tiles);
        zero_f32_kernel<<<1, 256, 0, stream>>>(stats, 256);
        col_stats_kernel<<<256, HID, 0, stream>>>(U, stats, stats + HID, N_NODES);
        finalize_stats_kernel<<<1, HID, 0, stream>>>(stats, stats + HID, 1.f / N_NODES, mi);
        bn_relu_f16_kernel<<<nh_blocks, TPB, 0, stream>>>(
            U, mi, mlp_bn_g + (size_t)l * HID, mlp_bn_b + (size_t)l * HID, V16);

        // MLP linear 2
        node_gemm_kernel<<<(node_tiles + 3) / 4, 128, 0, stream>>>(
            V16, w2p + (size_t)l * mlp_elems, mlp_b2 + (size_t)l * HID, Wb, node_tiles);

        // ApplyNodeFunc BN + ReLU
        zero_f32_kernel<<<1, 256, 0, stream>>>(stats, 256);
        col_stats_kernel<<<256, HID, 0, stream>>>(Wb, stats, stats + HID, N_NODES);
        finalize_stats_kernel<<<1, HID, 0, stream>>>(stats, stats + HID, 1.f / N_NODES, mi);
        bn_relu_f32_kernel<<<nh_blocks, TPB, 0, stream>>>(
            Wb, mi, app_bn_g + (size_t)l * HID, app_bn_b + (size_t)l * HID, Y1);

        // GINLayer BN + ReLU + residual
        zero_f32_kernel<<<1, 256, 0, stream>>>(stats, 256);
        col_stats_kernel<<<256, HID, 0, stream>>>(Y1, stats, stats + HID, N_NODES);
        finalize_stats_kernel<<<1, HID, 0, stream>>>(stats, stats + HID, 1.f / N_NODES, mi);
        gin_update_kernel<<<nh_blocks, TPB, 0, stream>>>(
            Y1, mi, gin_bn_g + (size_t)l * HID, gin_bn_b + (size_t)l * HID, X, Xh);

        // edge head l+1 (accumulate into score)
        edge_predict_kernel<<<edge_blocks, 128, 0, stream>>>(
            Xh, src, dst, p1p + (size_t)(l + 1) * pred_elems,
            pred_b1 + (size_t)(l + 1) * HID, pred_w2 + (size_t)(l + 1) * 2 * HID,
            pred_b2 + (size_t)(l + 1) * 2, out, /*accumulate=*/1);
    }
}
